// GATRes_60756607369580
// MI455X (gfx1250) — matrix operand order repaired and checked by reading, not compiled, not run on hardware
//
#include <hip/hip_runtime.h>

typedef unsigned int u32;
typedef unsigned long long u64;
typedef __bf16 bf16;
typedef bf16  v16bf __attribute__((ext_vector_type(16)));
typedef float v8f   __attribute__((ext_vector_type(8)));

union FragBf { u32 u[8]; v16bf v; };
union FragF  { float f[8]; v8f v; };

#define AS_STRIDE 34   // 32 + 2 bf16 pad -> 17-dword row stride, conflict-free

__device__ __forceinline__ u32 enc_f32(float f) {
    u32 u = __float_as_uint(f);
    return (u & 0x80000000u) ? ~u : (u | 0x80000000u);
}
__device__ __forceinline__ float dec_f32(u32 u) {
    u32 b = (u & 0x80000000u) ? (u ^ 0x80000000u) : ~u;
    return __uint_as_float(b);
}

// Async global->LDS dword copy (CDNA5 ASYNCcnt path; skips VGPR round trip).
__device__ __forceinline__ void async_copy_b32(const void* gsrc, void* lds_dst) {
    u32 ldsOff = (u32)(uintptr_t)lds_dst;          // flat LDS addr low 32 bits = LDS offset
    u64 ga = (u64)(uintptr_t)gsrc;
    asm volatile("global_load_async_to_lds_b32 %0, %1, off"
                 :: "v"(ldsOff), "v"(ga) : "memory");
}
__device__ __forceinline__ void async_wait0() {
    asm volatile("s_wait_asynccnt 0" ::: "memory");
}
__device__ __forceinline__ void async_wait1() {    // allow 1 copy in flight
    asm volatile("s_wait_asynccnt 1" ::: "memory");
}

// ---------------------------------------------------------------------------
// Tiled WMMA GEMM: C[M,Nc] = A[M,K](bf16) @ Wt^T  (+bias), Wt stored [Nc,K] bf16.
// Block = 256 thr = 8 waves; block tile 16(M) x 128(N); one 16x16 tile / wave.
// A tiles double-buffered in LDS via async global->LDS copies so the copy of
// tile k+1 overlaps the WMMA + weight-fragment loads of tile k.
// ---------------------------------------------------------------------------
template<bool OUT_BF16>
__global__ void wmma_gemm_kernel(const bf16* __restrict__ A,
                                 const bf16* __restrict__ Wt,
                                 const float* __restrict__ bias,
                                 void* __restrict__ Cout,
                                 int M, int K, int Nc) {
    __shared__ bf16 As[2][16 * AS_STRIDE];
    const int tid  = threadIdx.x;
    const int wave = tid >> 5, lane = tid & 31;
    const int lm = lane & 15, kh = lane >> 4;
    const int rowBase = blockIdx.x * 16;
    const int colBase = blockIdx.y * 128 + wave * 16;
    const int sm = tid >> 4, skp = tid & 15;       // staging coords (16x16 dwords)

    // prefetch this wave's weight panel (reused each M-block, L2 resident)
    __builtin_prefetch(Wt + (size_t)(colBase + lm) * K, 0, 3);

    FragF c;
#pragma unroll
    for (int r = 0; r < 8; ++r) c.f[r] = 0.f;

    const bf16* aRow = A + (size_t)(rowBase + sm) * K + 2 * skp;

    // prologue: stage tile 0 into buffer 0
    async_copy_b32(aRow, (char*)&As[0][sm * AS_STRIDE] + 4 * skp);

    int k0 = 0;
    for (; k0 + 32 < K; k0 += 32) {
        const int cur = (k0 >> 5) & 1;
        // stage tile k0+32 into the other buffer; overlaps compute below
        async_copy_b32(aRow + (k0 + 32), (char*)&As[cur ^ 1][sm * AS_STRIDE] + 4 * skp);
        async_wait1();                               // tile k0 resident (in-order)
        __syncthreads();

        FragBf a, b;
        const u32* arow = (const u32*)(&As[cur][lm * AS_STRIDE]);
        const u32* brow = (const u32*)(Wt + (size_t)(colBase + lm) * K + k0);
#pragma unroll
        for (int j = 0; j < 8; ++j) {
            int ka = (j < 4) ? (2 * j + 8 * kh) : (16 + 2 * (j - 4) + 8 * kh);
            int kb = (j < 4) ? (2 * j + 8 * kh) : (16 + 2 * (j - 4) + 8 * kh);
            a.u[j] = arow[ka >> 1];
            b.u[j] = brow[kb >> 1];
        }
        c.v = __builtin_amdgcn_wmma_f32_16x16x32_bf16(
                  false, a.v, false, b.v, (short)0, c.v, false, false);
        __syncthreads();                             // done reading before overwrite
    }
    {   // epilogue: last tile
        const int cur = (k0 >> 5) & 1;
        async_wait0();
        __syncthreads();
        FragBf a, b;
        const u32* arow = (const u32*)(&As[cur][lm * AS_STRIDE]);
        const u32* brow = (const u32*)(Wt + (size_t)(colBase + lm) * K + k0);
#pragma unroll
        for (int j = 0; j < 8; ++j) {
            int ka = (j < 4) ? (2 * j + 8 * kh) : (16 + 2 * (j - 4) + 8 * kh);
            int kb = (j < 4) ? (2 * j + 8 * kh) : (16 + 2 * (j - 4) + 8 * kh);
            a.u[j] = arow[ka >> 1];
            b.u[j] = brow[kb >> 1];
        }
        c.v = __builtin_amdgcn_wmma_f32_16x16x32_bf16(
                  false, a.v, false, b.v, (short)0, c.v, false, false);
    }

    const int col = colBase + lm;
    const float bs = bias ? bias[col] : 0.f;
    if (OUT_BF16) {
        bf16* C = (bf16*)Cout;
#pragma unroll
        for (int r = 0; r < 8; ++r) {
            int m = rowBase + r + 8 * kh;
            C[(size_t)m * Nc + col] = (bf16)(c.f[r] + bs);
        }
    } else {
        float* C = (float*)Cout;
#pragma unroll
        for (int r = 0; r < 8; ++r) {
            int m = rowBase + r + 8 * kh;
            C[(size_t)m * Nc + col] = c.f[r] + bs;
        }
    }
}

// ---------------------------------------------------------------------------
// Per-(graph,head) dense attention: S=64, DH=32.  Q·K^T via one K=32 WMMA per
// 16x16 tile; softmax in LDS; (softmax)·V via two K=32 WMMA steps.
// ---------------------------------------------------------------------------
__global__ void attn_kernel(const bf16* __restrict__ q,
                            const bf16* __restrict__ k,
                            const bf16* __restrict__ v,
                            bf16* __restrict__ obf) {
    __shared__ bf16 qs[64 * AS_STRIDE];
    __shared__ bf16 ks[64 * AS_STRIDE];
    __shared__ bf16 vt[32 * 66];        // v transposed: [dh][key]
    __shared__ float sc[64 * 65];       // scores / exp
    __shared__ bf16 ab[64 * 66];        // softmax probs, bf16

    const int b = blockIdx.x, h = blockIdx.y;
    const int tid = threadIdx.x, lane = tid & 31, wave = tid >> 5;
    const int lm = lane & 15, kh = lane >> 4;

    for (int t = tid; t < 1024; t += 256) {      // q,k tiles: async dword copies
        int row = t >> 4, p = t & 15;
        size_t g = (size_t)(b * 64 + row) * 128 + h * 32 + 2 * p;
        async_copy_b32(q + g, (char*)&qs[row * AS_STRIDE] + 4 * p);
        async_copy_b32(k + g, (char*)&ks[row * AS_STRIDE] + 4 * p);
    }
    for (int t = tid; t < 2048; t += 256) {      // v transposed elementwise
        int key = t >> 5, d = t & 31;
        vt[d * 66 + key] = v[(size_t)(b * 64 + key) * 128 + h * 32 + d];
    }
    async_wait0();
    __syncthreads();

    const float scale = 0.17677669529663687f;    // 1/sqrt(32)
    for (int t = wave; t < 16; t += 8) {         // 16 score tiles, 2 per wave
        int mt = t >> 2, nt = t & 3;
        FragBf a, bb;
        const u32* arow = (const u32*)(&qs[(mt * 16 + lm) * AS_STRIDE]);
        const u32* brow = (const u32*)(&ks[(nt * 16 + lm) * AS_STRIDE]);
#pragma unroll
        for (int j = 0; j < 8; ++j) {
            int ka = (j < 4) ? (2 * j + 8 * kh) : (16 + 2 * (j - 4) + 8 * kh);
            int kb = (j < 4) ? (2 * j + 8 * kh) : (16 + 2 * (j - 4) + 8 * kh);
            a.u[j] = arow[ka >> 1];
            bb.u[j] = brow[kb >> 1];
        }
        FragF c;
#pragma unroll
        for (int r = 0; r < 8; ++r) c.f[r] = 0.f;
        c.v = __builtin_amdgcn_wmma_f32_16x16x32_bf16(
                  false, a.v, false, bb.v, (short)0, c.v, false, false);
#pragma unroll
        for (int r = 0; r < 8; ++r)
            sc[(mt * 16 + r + 8 * kh) * 65 + nt * 16 + lm] = c.f[r] * scale;
    }
    __syncthreads();

    if (tid < 64) {                              // row softmax -> bf16 probs
        float mx = -1e30f;
        for (int j = 0; j < 64; ++j) mx = fmaxf(mx, sc[tid * 65 + j]);
        float s = 0.f;
        for (int j = 0; j < 64; ++j) {
            float ex = __expf(sc[tid * 65 + j] - mx);
            s += ex;
            sc[tid * 65 + j] = ex;
        }
        float inv = 1.f / s;
        for (int j = 0; j < 64; ++j)
            ab[tid * 66 + j] = (bf16)(sc[tid * 65 + j] * inv);
    }
    __syncthreads();

    {   // o = probs(64x64) @ v(64x32): 8 tiles, one per wave, K = 64 (2 steps)
        int mt = wave >> 1, nt = wave & 1;
        FragF c;
#pragma unroll
        for (int r = 0; r < 8; ++r) c.f[r] = 0.f;
        for (int k0 = 0; k0 < 64; k0 += 32) {
            FragBf a, bb;
            const u32* arow = (const u32*)(&ab[(mt * 16 + lm) * 66 + k0]);
            const u32* brow = (const u32*)(&vt[(nt * 16 + lm) * 66 + k0]);
#pragma unroll
            for (int j = 0; j < 8; ++j) {
                int ka = (j < 4) ? (2 * j + 8 * kh) : (16 + 2 * (j - 4) + 8 * kh);
                int kb = (j < 4) ? (2 * j + 8 * kh) : (16 + 2 * (j - 4) + 8 * kh);
                a.u[j] = arow[ka >> 1];
                bb.u[j] = brow[kb >> 1];
            }
            c.v = __builtin_amdgcn_wmma_f32_16x16x32_bf16(
                      false, a.v, false, bb.v, (short)0, c.v, false, false);
        }
#pragma unroll
        for (int r = 0; r < 8; ++r) {
            int row = mt * 16 + r + 8 * kh;
            int d = nt * 16 + lm;
            obf[(size_t)(b * 64 + row) * 128 + h * 32 + d] = (bf16)c.f[r];
        }
    }
}

// ---------------------------------------------------------------------------
// Edge-wise GATv2 stages (H_ heads, per-head dim fixed = 128)
// ---------------------------------------------------------------------------
__global__ void edge_logit_kernel(const int* __restrict__ src, const int* __restrict__ dst,
                                  const float* __restrict__ ea,
                                  const bf16* __restrict__ xl, const bf16* __restrict__ xr,
                                  const float* __restrict__ We, const float* __restrict__ att,
                                  float* __restrict__ logit, u32* __restrict__ mxU,
                                  int Ecnt, int H_) {
    int wg = blockIdx.x * 8 + (threadIdx.x >> 5);
    int lane = threadIdx.x & 31;
    if (wg >= Ecnt * H_) return;
    int e = wg / H_, h = wg - e * H_;
    int s = src[e], d = dst[e];
    float e0 = ea[e * 3 + 0], e1 = ea[e * 3 + 1], e2 = ea[e * 3 + 2];
    size_t sb = (size_t)s * (H_ * 128) + h * 128;
    size_t db = (size_t)d * (H_ * 128) + h * 128;
    int W = H_ * 128;
    float acc = 0.f;
#pragma unroll
    for (int i = 0; i < 4; ++i) {
        int dd = lane + 32 * i;
        float ee = e0 * We[0 * W + h * 128 + dd] + e1 * We[1 * W + h * 128 + dd]
                 + e2 * We[2 * W + h * 128 + dd];
        float m = (float)xl[sb + dd] + (float)xr[db + dd] + ee;
        m = m > 0.f ? m : 0.2f * m;                       // leaky_relu 0.2
        acc += att[h * 128 + dd] * m;
    }
    for (int off = 16; off > 0; off >>= 1) acc += __shfl_xor(acc, off, 32);
    if (lane == 0) {
        logit[(size_t)e * H_ + h] = acc;
        atomicMax(&mxU[(size_t)d * H_ + h], enc_f32(acc));
    }
}

__global__ void edge_den_kernel(const int* __restrict__ dst, const float* __restrict__ logit,
                                const u32* __restrict__ mxU, float* __restrict__ den,
                                int Ecnt, int H_) {
    int t = blockIdx.x * 256 + threadIdx.x;
    if (t >= Ecnt * H_) return;
    int e = t / H_, h = t - e * H_;
    int d = dst[e];
    float mx = dec_f32(mxU[(size_t)d * H_ + h]);
    atomicAdd(&den[(size_t)d * H_ + h], __expf(logit[t] - mx));
}

__global__ void edge_scatter_kernel(const int* __restrict__ src, const int* __restrict__ dst,
                                    const bf16* __restrict__ xl,
                                    const float* __restrict__ logit, const u32* __restrict__ mxU,
                                    const float* __restrict__ den, float* __restrict__ accum,
                                    int Ecnt, int H_, float scale) {
    int e = blockIdx.x * 8 + (threadIdx.x >> 5);
    int lane = threadIdx.x & 31;
    if (e >= Ecnt) return;
    int s = src[e], d = dst[e];
    float alpha[4];
    for (int h = 0; h < H_; ++h) {
        float mx = dec_f32(mxU[(size_t)d * H_ + h]);
        alpha[h] = __expf(logit[(size_t)e * H_ + h] - mx) / den[(size_t)d * H_ + h];
    }
    size_t sb = (size_t)s * (H_ * 128);
    size_t ob = (size_t)d * 128;
#pragma unroll
    for (int i = 0; i < 4; ++i) {
        int dd = lane + 32 * i;
        float acc = 0.f;
        for (int h = 0; h < H_; ++h) acc += alpha[h] * (float)xl[sb + h * 128 + dd];
        atomicAdd(&accum[ob + dd], acc * scale);
    }
}

// ---------------------------------------------------------------------------
// Small utility kernels
// ---------------------------------------------------------------------------
__global__ void init_seg_kernel(u32* mxU, float* den, int n) {
    int t = blockIdx.x * 256 + threadIdx.x;
    if (t >= n) return;
    mxU[t] = 0x007FFFFFu;   // enc(-inf)
    den[t] = 0.f;
}
__global__ void init_accum_kernel(float* accum, const float* bias, const float* hprev, int n) {
    int t = blockIdx.x * 256 + threadIdx.x;
    if (t >= n) return;
    accum[t] = bias[t & 127] + (hprev ? hprev[t] : 0.f);
}
__global__ void scale_copy_kernel(const float* in, float* out, float s, int n) {
    int t = blockIdx.x * 256 + threadIdx.x;
    if (t >= n) return;
    out[t] = in[t] * s;
}
__global__ void convert_bf_kernel(const float* in, bf16* out, int n, int relu) {
    int t = blockIdx.x * 256 + threadIdx.x;
    if (t >= n) return;
    float v = in[t];
    if (relu) v = v > 0.f ? v : 0.f;
    out[t] = (bf16)v;
}
__global__ void xpad_kernel(const float* x, bf16* out, int n32) {   // pad 9 -> 32
    int t = blockIdx.x * 256 + threadIdx.x;
    if (t >= n32) return;
    int n = t >> 5, j = t & 31;
    out[t] = (bf16)(j < 9 ? x[n * 9 + j] : 0.f);
}
__global__ void wt_kernel(const float* W, bf16* Wt, int K, int Kpad, int Nc) {
    int t = blockIdx.x * 256 + threadIdx.x;
    if (t >= Nc * Kpad) return;
    int n = t / Kpad, kp = t - n * Kpad;
    Wt[t] = (bf16)(kp < K ? W[kp * Nc + n] : 0.f);     // transpose + zero-pad
}
__global__ void final_kernel(const float* h, const float* op, float* out, int n) {
    int t = blockIdx.x * 256 + threadIdx.x;
    if (t >= n) return;
    out[t] = 0.5f * (h[t] + op[t]);
}

// ---------------------------------------------------------------------------
extern "C" void kernel_launch(void* const* d_in, const int* in_sizes, int n_in,
                              void* d_out, int out_size, void* d_ws, size_t ws_size,
                              hipStream_t stream) {
    constexpr int N_ = 65536, E_ = 262144, B_ = 1024, H_ = 4;
    constexpr size_t MB = 1u << 20;

    const float* x      = (const float*)d_in[0];
    const int*   src    = (const int*)d_in[1];
    const int*   dst    = src + E_;
    const float* ea     = (const float*)d_in[2];
    const float* g0_Wl  = (const float*)d_in[4];
    const float* g0_Wr  = (const float*)d_in[5];
    const float* g0_bl  = (const float*)d_in[6];
    const float* g0_br  = (const float*)d_in[7];
    const float* g0_We  = (const float*)d_in[8];
    const float* g0_att = (const float*)d_in[9];
    const float* g0_bias= (const float*)d_in[10];
    const float* gWl    = (const float*)d_in[11];
    const float* gWr    = (const float*)d_in[12];
    const float* gbl    = (const float*)d_in[13];
    const float* gbr    = (const float*)d_in[14];
    const float* gWe    = (const float*)d_in[15];
    const float* gatt   = (const float*)d_in[16];
    const float* gbias  = (const float*)d_in[17];
    const float* Wq     = (const float*)d_in[18];
    const float* Wk     = (const float*)d_in[19];
    const float* Wv     = (const float*)d_in[20];
    const float* bq     = (const float*)d_in[21];
    const float* bk     = (const float*)d_in[22];
    const float* bv     = (const float*)d_in[23];
    const float* Wo     = (const float*)d_in[24];
    const float* bo     = (const float*)d_in[25];

    char* ws = (char*)d_ws;
    float* h     = (float*)(ws);                 // 32 MB
    bf16*  hbf   = (bf16*)(ws + 32 * MB);        // 16 MB
    bf16*  xlbf  = (bf16*)(ws + 48 * MB);        // 64 MB
    bf16*  xrbf  = (bf16*)(ws + 112 * MB);       // 64 MB
    float* accum = (float*)(ws + 176 * MB);      // 32 MB
    float* logit = (float*)(ws + 208 * MB);      //  4 MB
    u32*   mxU   = (u32*)(ws + 212 * MB);        //  1 MB
    float* den   = (float*)(ws + 213 * MB);      //  1 MB
    bf16*  xpad  = (bf16*)(ws + 214 * MB);       //  4 MB
    char*  wb    = ws + 218 * MB;
    bf16* W0lT = (bf16*)(wb);                    // 128x32
    bf16* W0rT = (bf16*)(wb + 8 * 1024);
    bf16* WlT[3], *WrT[3];
    for (int i = 0; i < 3; ++i) {
        WlT[i] = (bf16*)(wb + 16 * 1024 + (size_t)i * 128 * 1024);
        WrT[i] = (bf16*)(wb + 16 * 1024 + (384 + (size_t)i * 128) * 1024);
    }
    bf16* WqT = (bf16*)(wb + 800 * 1024);
    bf16* WkT = (bf16*)(wb + 832 * 1024);
    bf16* WvT = (bf16*)(wb + 864 * 1024);
    bf16* WoT = (bf16*)(wb + 896 * 1024);
    // attention-phase aliases (GNN scratch no longer live)
    bf16*  qb    = xlbf;
    bf16*  kb    = xlbf + (size_t)N_ * 128;
    bf16*  vb    = xlbf + (size_t)N_ * 128 * 2;
    bf16*  obf   = xlbf + (size_t)N_ * 128 * 3;
    float* oproj = (float*)xrbf;

    // ---- weight conversion (transpose to [Nc,K] bf16, zero-pad K) ----
    wt_kernel<<<(128 * 32 + 255) / 256, 256, 0, stream>>>(g0_Wl, W0lT, 9, 32, 128);
    wt_kernel<<<(128 * 32 + 255) / 256, 256, 0, stream>>>(g0_Wr, W0rT, 9, 32, 128);
    for (int i = 0; i < 3; ++i) {
        wt_kernel<<<(512 * 128 + 255) / 256, 256, 0, stream>>>(gWl + (size_t)i * 128 * 512, WlT[i], 128, 128, 512);
        wt_kernel<<<(512 * 128 + 255) / 256, 256, 0, stream>>>(gWr + (size_t)i * 128 * 512, WrT[i], 128, 128, 512);
    }
    wt_kernel<<<(128 * 128 + 255) / 256, 256, 0, stream>>>(Wq, WqT, 128, 128, 128);
    wt_kernel<<<(128 * 128 + 255) / 256, 256, 0, stream>>>(Wk, WkT, 128, 128, 128);
    wt_kernel<<<(128 * 128 + 255) / 256, 256, 0, stream>>>(Wv, WvT, 128, 128, 128);
    wt_kernel<<<(128 * 128 + 255) / 256, 256, 0, stream>>>(Wo, WoT, 128, 128, 128);
    xpad_kernel<<<N_ * 32 / 256, 256, 0, stream>>>(x, xpad, N_ * 32);

    // ---- layer 0: GATv2, H=1, concat ----
    wmma_gemm_kernel<true><<<dim3(N_ / 16, 1), 256, 0, stream>>>(xpad, W0lT, g0_bl, xlbf, N_, 32, 128);
    wmma_gemm_kernel<true><<<dim3(N_ / 16, 1), 256, 0, stream>>>(xpad, W0rT, g0_br, xrbf, N_, 32, 128);
    init_seg_kernel<<<N_ / 256, 256, 0, stream>>>(mxU, den, N_);
    init_accum_kernel<<<N_ * 128 / 256, 256, 0, stream>>>(accum, g0_bias, nullptr, N_ * 128);
    edge_logit_kernel<<<E_ / 8, 256, 0, stream>>>(src, dst, ea, xlbf, xrbf, g0_We, g0_att, logit, mxU, E_, 1);
    edge_den_kernel<<<E_ / 256, 256, 0, stream>>>(dst, logit, mxU, den, E_, 1);
    edge_scatter_kernel<<<E_ / 8, 256, 0, stream>>>(src, dst, xlbf, logit, mxU, den, accum, E_, 1, 1.0f);
    scale_copy_kernel<<<N_ * 128 / 256, 256, 0, stream>>>(accum, h, 1.0f, N_ * 128);

    // ---- layers 1..3: GATv2, H=4, mean + residual ----
    for (int i = 0; i < 3; ++i) {
        convert_bf_kernel<<<N_ * 128 / 256, 256, 0, stream>>>(h, hbf, N_ * 128, 1);   // relu(h)
        wmma_gemm_kernel<true><<<dim3(N_ / 16, 4), 256, 0, stream>>>(hbf, WlT[i], gbl + (size_t)i * 512, xlbf, N_, 128, 512);
        wmma_gemm_kernel<true><<<dim3(N_ / 16, 4), 256, 0, stream>>>(hbf, WrT[i], gbr + (size_t)i * 512, xrbf, N_, 128, 512);
        init_seg_kernel<<<N_ * H_ / 256, 256, 0, stream>>>(mxU, den, N_ * H_);
        init_accum_kernel<<<N_ * 128 / 256, 256, 0, stream>>>(accum, gbias + (size_t)i * 128, h, N_ * 128);
        edge_logit_kernel<<<E_ * H_ / 8, 256, 0, stream>>>(src, dst, ea, xlbf, xrbf,
            gWe + (size_t)i * 3 * 512, gatt + (size_t)i * 4 * 128, logit, mxU, E_, H_);
        edge_den_kernel<<<E_ * H_ / 256, 256, 0, stream>>>(dst, logit, mxU, den, E_, H_);
        edge_scatter_kernel<<<E_ / 8, 256, 0, stream>>>(src, dst, xlbf, logit, mxU, den, accum, E_, H_, 0.25f);
        scale_copy_kernel<<<N_ * 128 / 256, 256, 0, stream>>>(accum, h, 0.5f, N_ * 128);
    }

    // ---- dense per-graph attention ----
    convert_bf_kernel<<<N_ * 128 / 256, 256, 0, stream>>>(h, hbf, N_ * 128, 0);
    wmma_gemm_kernel<true><<<dim3(N_ / 16, 1), 256, 0, stream>>>(hbf, WqT, bq, qb, N_, 128, 128);
    wmma_gemm_kernel<true><<<dim3(N_ / 16, 1), 256, 0, stream>>>(hbf, WkT, bk, kb, N_, 128, 128);
    wmma_gemm_kernel<true><<<dim3(N_ / 16, 1), 256, 0, stream>>>(hbf, WvT, bv, vb, N_, 128, 128);
    attn_kernel<<<dim3(B_, H_), 256, 0, stream>>>(qb, kb, vb, obf);
    wmma_gemm_kernel<false><<<dim3(N_ / 16, 1), 256, 0, stream>>>(obf, WoT, bo, oproj, N_, 128, 128);
    final_kernel<<<N_ * 128 / 256, 256, 0, stream>>>(h, oproj, (float*)d_out, N_ * 128);
    (void)in_sizes; (void)n_in; (void)out_size; (void)ws_size;
}
